// YOLOLayer_19550691131820
// MI455X (gfx1250) — matrix operand order repaired and checked by reading, not compiled, not run on hardware
//
#include <hip/hip_runtime.h>

// ---------------- problem constants ----------------
#define NBATCH 16
#define NANCH  3
#define NCH    85          // 5 + 80 classes
#define NOUT   255         // NANCH * NCH
#define NG     76
#define NS     (NG*NG)     // 5776
#define KDIM   256
#define NBOX   50
#define NPAIR  (NBATCH*NBOX)  // 800

typedef __attribute__((ext_vector_type(16))) __bf16 v16bf;
typedef __attribute__((ext_vector_type(8)))  __bf16 v8bf;
typedef __attribute__((ext_vector_type(2)))  __bf16 v2bf;
typedef __attribute__((ext_vector_type(8)))  float  v8f;

// async global->LDS path (CDNA5), guarded for toolchain portability
#if defined(__has_builtin)
#if __has_builtin(__builtin_amdgcn_global_load_async_to_lds_b64)
#define GEMM_ASYNC 1
#endif
#if __has_builtin(__builtin_amdgcn_cvt_pk_bf16_f32)
#define HAVE_PK_BF16 1
#endif
#endif

// exact pointee type expected by the async builtin (from hipcc diagnostic):
// '__attribute__((__vector_size__(2 * sizeof(int)))) int' in AS1 (src) / AS3 (dst)
typedef int v2i32 __attribute__((vector_size(8)));
typedef __attribute__((address_space(1))) v2i32* as1_v2i;
typedef __attribute__((address_space(3))) v2i32* as3_v2i;

// ALL_ANCHORS = ANCHORS_PX / 8  (first 3 are the active anchors)
__constant__ float c_aw[9] = {1.25f, 2.0f, 4.125f, 3.75f, 7.75f, 7.375f, 14.5f, 19.5f, 46.625f};
__constant__ float c_ah[9] = {1.625f, 3.75f, 2.875f, 7.625f, 5.625f, 14.875f, 11.25f, 24.75f, 40.75f};

// ---------------- workspace layout (float units) ----------------
#define WS_ACC   0                  // [0]=xy [1]=wh [2]=obj [3]=cls (8 floats reserved)
#define WS_PI    8                  // 6 int arrays x 800
#define WS_PF    (8 + 6*NPAIR)      // 9 float arrays x 800
#define WS_WBF   16384              // packed bf16 W fragments: 65536 bf16 = 32768 floats
#define WS_XPOST 49152              // 16*255*5776 floats (~94 MB)

#define PI_ASSIGN 0
#define PI_A      1
#define PI_GI     2
#define PI_GJ     3
#define PI_CLS    4
#define PI_VALID  5

#define PF_TX 0
#define PF_TY 1
#define PF_TW 2
#define PF_TH 3
#define PF_SV 4
#define PF_GX 5
#define PF_GY 6
#define PF_GW 7
#define PF_GH 8

__device__ __forceinline__ float slogf_(float p) {
  return fmaxf(__logf(fmaxf(p, 1e-12f)), -100.0f);
}
__device__ __forceinline__ float sigm(float x) { return 1.0f / (1.0f + __expf(-x)); }
__device__ __forceinline__ __bf16 f2bf(float f) { return (__bf16)f; }   // native v_cvt

__device__ __forceinline__ void async_wait_keep1() {
#if defined(GEMM_ASYNC)
#if __has_builtin(__builtin_amdgcn_s_wait_asynccnt)
  __builtin_amdgcn_s_wait_asynccnt(1);
#else
  asm volatile("s_wait_asynccnt 0x1" ::: "memory");
#endif
#endif
}
__device__ __forceinline__ void async_wait_all() {
#if defined(GEMM_ASYNC)
#if __has_builtin(__builtin_amdgcn_s_wait_asynccnt)
  __builtin_amdgcn_s_wait_asynccnt(0);
#else
  asm volatile("s_wait_asynccnt 0x0" ::: "memory");
#endif
#endif
}

// ---------------- accumulator init ----------------
__global__ void init_kernel(float* ws) {
  if (threadIdx.x < 8) ws[threadIdx.x] = 0.0f;
}

// ---------------- pack W into bf16 WMMA A-fragment layout ----------------
// chunk index = (oT*8 + step)*32 + lane ; each chunk = 16 bf16 (one lane's A frag)
__global__ void packW_kernel(const float* __restrict__ Wm, __bf16* __restrict__ wbf) {
  const int idx = blockIdx.x * blockDim.x + threadIdx.x;   // 16*8*32 = 4096
  if (idx >= 4096) return;
  const int lane = idx & 31;
  const int stp  = (idx >> 5) & 7;
  const int oT   = idx >> 8;
  const int hi = lane >> 4, lo = lane & 15;
  const int m  = oT * 16 + lo;
  const int k0 = stp * 32;
  __bf16* dst = wbf + (size_t)idx * 16;
  #pragma unroll
  for (int j = 0; j < 8; ++j) {
    const int kA = k0 + 8 * hi + j;
    const int kB = k0 + 16 + 8 * hi + j;
    const float va = (m < NOUT) ? Wm[m * KDIM + kA] : 0.0f;
    const float vb = (m < NOUT) ? Wm[m * KDIM + kB] : 0.0f;
    dst[j]     = f2bf(va);
    dst[8 + j] = f2bf(vb);
  }
}

// ---------------- GEMM + bias + selective sigmoid (WMMA bf16, LDS-staged) ----
// Block = one (b, sTile): 256 output rows x 16 spatial columns.
// 8 waves x 2 output tiles each. Pipeline per 32-K step:
//   async f32 stage -> LDS (double-buffered)  -> block-wide packed cvt into a
//   transposed bf16 tile -> each lane reads its B fragment as 2x ds_load_b128.
#define FROW 18   // f32 staging row pitch (dwords): conflict-free, 8B pairs
#define HROW 40   // bf16 tile row pitch (halves): 80B rows, 16B-aligned chunks

__device__ __forceinline__ void stage_step(const float* __restrict__ xb, int n0,
                                           int k0, float* lbuf, int tid) {
  const int k = tid >> 3;          // 0..31 : K row within tile
  const int p = (tid & 7) << 1;    // 0,2,..,14 : column pair
  const float* gp = xb + (size_t)(k0 + k) * NS + n0 + p;
  float* lp = lbuf + k * FROW + p;
#if defined(GEMM_ASYNC)
  __builtin_amdgcn_global_load_async_to_lds_b64((as1_v2i)gp, (as3_v2i)lp, 0, 0);
#else
  float2 v = *(const float2*)gp;
  *(float2*)lp = v;
#endif
}

__global__ void __launch_bounds__(256) gemm_wmma_kernel(const float* __restrict__ xin,
                                                        const __bf16* __restrict__ wbf,
                                                        const float* __restrict__ bias,
                                                        float* __restrict__ xpost) {
  __shared__ float  fbuf[2][32 * FROW];
  __shared__ __bf16 hbuf[16 * HROW];

  const int tid  = threadIdx.x;
  const int wave = tid >> 5;
  const int lane = tid & 31;
  const int hi = lane >> 4;
  const int lo = lane & 15;

  const int b  = blockIdx.x / 361;
  const int sT = blockIdx.x % 361;
  const int n0 = sT * 16;

  const int oT0 = wave * 2;
  const int oT1 = wave * 2 + 1;

  // convert-pass indices: thread -> (column n, K pair)
  const int cn  = tid & 15;
  const int ckp = tid >> 4;     // 0..15

  const float* xb = xin + (size_t)b * KDIM * NS;
  const v16bf* wfrag = (const v16bf*)wbf;

  v8f acc0 = {};
  v8f acc1 = {};

  stage_step(xb, n0, 0, &fbuf[0][0], tid);

  #pragma unroll
  for (int step = 0; step < 8; ++step) {
    const int k0 = step * 32;
    if (step < 7) {
      stage_step(xb, n0, k0 + 32, &fbuf[(step + 1) & 1][0], tid);
      async_wait_keep1();     // oldest (current buffer) copy complete
    } else {
      async_wait_all();
    }
    __syncthreads();          // fbuf[cur] visible to all

    // block-wide convert: 512 f32 -> transposed bf16 tile, 2 values/thread
    {
      const float* fb = &fbuf[step & 1][0];
      const float f0 = fb[(2 * ckp)     * FROW + cn];
      const float f1 = fb[(2 * ckp + 1) * FROW + cn];
      __bf16* dst = &hbuf[cn * HROW + 2 * ckp];
#if defined(HAVE_PK_BF16)
      *(v2bf*)dst = __builtin_amdgcn_cvt_pk_bf16_f32(f0, f1);
#else
      dst[0] = f2bf(f0);
      dst[1] = f2bf(f1);
#endif
    }
    __syncthreads();          // hbuf ready

    // B fragment: row lo, halves K = k0 + 8*hi + (0..7) and k0 + 16 + 8*hi + (0..7) -> two 16B LDS loads
    const __bf16* hp = &hbuf[lo * HROW + 8 * hi];
    const v8bf lob = *(const v8bf*)hp;
    const v8bf hib = *(const v8bf*)(hp + 16);
    const v16bf Bv = __builtin_shufflevector(lob, hib,
        0, 1, 2, 3, 4, 5, 6, 7, 8, 9, 10, 11, 12, 13, 14, 15);

    // A fragments: pre-packed, one 32B chunk per lane per tile
    const v16bf A0 = wfrag[(size_t)(oT0 * 8 + step) * 32 + lane];
    const v16bf A1 = wfrag[(size_t)(oT1 * 8 + step) * 32 + lane];

    acc0 = __builtin_amdgcn_wmma_f32_16x16x32_bf16(false, A0, false, Bv,
                                                   (short)0, acc0, false, false);
    acc1 = __builtin_amdgcn_wmma_f32_16x16x32_bf16(false, A1, false, Bv,
                                                   (short)0, acc1, false, false);
    __syncthreads();          // protect hbuf (and fbuf) before next step
  }

  // Epilogue: D VGPR r, lane -> row r + 8*hi, col lane&15
  const int s = n0 + lo;
  #pragma unroll
  for (int r = 0; r < 8; ++r) {
    const int o0 = oT0 * 16 + r + 8 * hi;
    {
      float v = acc0[r] + bias[o0];
      const int c = o0 % NCH;
      if (c != 2 && c != 3) v = sigm(v);
      xpost[((size_t)b * NOUT + o0) * NS + s] = v;
    }
    const int o1 = oT1 * 16 + r + 8 * hi;
    if (o1 < NOUT) {
      float v = acc1[r] + bias[o1];
      const int c = o1 % NCH;
      if (c != 2 && c != 3) v = sigm(v);
      xpost[((size_t)b * NOUT + o1) * NS + s] = v;
    }
  }
}

// ---------------- per-GT preprocessing ----------------
__global__ void prep_kernel(const float* __restrict__ labels, float* ws) {
  const int idx = blockIdx.x * blockDim.x + threadIdx.x;
  if (idx >= NPAIR) return;
  const float* L = labels + (size_t)idx * 5;
  float cls = L[0], cx = L[1], cy = L[2], w = L[3], h = L[4];
  int valid = (cls + cx + cy + w + h) > 0.0f;
  float gx_ = cx * NG, gy_ = cy * NG, gw_ = w * NG, gh_ = h * NG;
  int gi = (int)floorf(gx_), gj = (int)floorf(gy_);

  float bestR = -1.0f; int best = 0;        // first-max == jnp.argmax
  for (int k = 0; k < 9; ++k) {
    float inter = fminf(gw_, c_aw[k]) * fminf(gh_, c_ah[k]);
    float uni   = gw_ * gh_ + c_aw[k] * c_ah[k] - inter;
    float rto   = inter / uni;
    if (rto > bestR) { bestR = rto; best = k; }
  }
  int a      = (best <= 2) ? best : -1;     // anchor window [0,2]
  int assign = valid && (a >= 0);
  int asafe  = (a < 0) ? 0 : a;

  int*   pi = (int*)(ws + WS_PI);
  float* pf = ws + WS_PF;
  pi[PI_ASSIGN * NPAIR + idx] = assign;
  pi[PI_A      * NPAIR + idx] = asafe;
  pi[PI_GI     * NPAIR + idx] = gi;
  pi[PI_GJ     * NPAIR + idx] = gj;
  pi[PI_CLS    * NPAIR + idx] = (int)cls;
  pi[PI_VALID  * NPAIR + idx] = valid;
  pf[PF_TX * NPAIR + idx] = gx_ - floorf(gx_);
  pf[PF_TY * NPAIR + idx] = gy_ - floorf(gy_);
  pf[PF_TW * NPAIR + idx] = logf(gw_ / c_aw[asafe] + 1e-16f);
  pf[PF_TH * NPAIR + idx] = logf(gh_ / c_ah[asafe] + 1e-16f);
  pf[PF_SV * NPAIR + idx] = 2.0f - gw_ * gh_ / (float)NS;
  pf[PF_GX * NPAIR + idx] = gx_;
  pf[PF_GY * NPAIR + idx] = gy_;
  pf[PF_GW * NPAIR + idx] = gw_;
  pf[PF_GH * NPAIR + idx] = gh_;
}

// ---------------- dense objectness BCE over all cells ----------------
__global__ void __launch_bounds__(256) obj_kernel(const float* __restrict__ xpost, float* ws) {
  __shared__ float sgx[NBOX], sgy[NBOX], sgw[NBOX], sgh[NBOX];
  __shared__ int   sval[NBOX], sasg[NBOX], saA[NBOX], sgi[NBOX], sgj[NBOX];
  __shared__ float red[256];

  const int b = blockIdx.y;
  const int* pi = (const int*)(ws + WS_PI);
  const float* pf = ws + WS_PF;
  if (threadIdx.x < NBOX) {
    int idx = b * NBOX + threadIdx.x;
    sval[threadIdx.x] = pi[PI_VALID  * NPAIR + idx];
    sasg[threadIdx.x] = pi[PI_ASSIGN * NPAIR + idx];
    saA[threadIdx.x]  = pi[PI_A      * NPAIR + idx];
    sgi[threadIdx.x]  = pi[PI_GI     * NPAIR + idx];
    sgj[threadIdx.x]  = pi[PI_GJ     * NPAIR + idx];
    sgx[threadIdx.x]  = pf[PF_GX * NPAIR + idx];
    sgy[threadIdx.x]  = pf[PF_GY * NPAIR + idx];
    sgw[threadIdx.x]  = pf[PF_GW * NPAIR + idx];
    sgh[threadIdx.x]  = pf[PF_GH * NPAIR + idx];
  }
  __syncthreads();

  const int cell = blockIdx.x * blockDim.x + threadIdx.x;   // within one batch
  float contrib = 0.0f;
  if (cell < NANCH * NS) {
    const int a = cell / NS;
    const int s = cell % NS;
    const int i = s % NG;
    const int j = s / NG;
    const size_t cb = ((size_t)b * NOUT + a * NCH) * NS + s;
    const float px   = xpost[cb];
    const float py   = xpost[cb + (size_t)NS];
    const float bw   = __expf(xpost[cb + 2 * (size_t)NS]) * c_aw[a];
    const float bh   = __expf(xpost[cb + 3 * (size_t)NS]) * c_ah[a];
    const float conf = xpost[cb + 4 * (size_t)NS];
    const float bx = px + (float)i;
    const float by = py + (float)j;
    const float parea = bw * bh;

    bool hasb = false, assigned = false;
    float best = 0.0f;
    #pragma unroll 1
    for (int nn = 0; nn < NBOX; ++nn) {
      if (sval[nn]) {
        hasb = true;
        float gx = sgx[nn], gy = sgy[nn], gw = sgw[nn], gh = sgh[nn];
        float tlx = fmaxf(bx - bw * 0.5f, gx - gw * 0.5f);
        float tly = fmaxf(by - bh * 0.5f, gy - gh * 0.5f);
        float brx = fminf(bx + bw * 0.5f, gx + gw * 0.5f);
        float bry = fminf(by + bh * 0.5f, gy + gh * 0.5f);
        float inter = ((tlx < brx) && (tly < bry)) ? (brx - tlx) * (bry - tly) : 0.0f;
        float iou = inter / (parea + gw * gh - inter);
        best = fmaxf(best, iou);
      }
      if (sasg[nn] && saA[nn] == a && sgi[nn] == i && sgj[nn] == j) assigned = true;
    }
    bool objm = assigned || (!hasb) || (best <= 0.7f);
    if (objm) contrib = assigned ? -slogf_(conf) : -slogf_(1.0f - conf);
  }

  red[threadIdx.x] = contrib;
  __syncthreads();
  for (int off = 128; off > 0; off >>= 1) {
    if (threadIdx.x < off) red[threadIdx.x] += red[threadIdx.x + off];
    __syncthreads();
  }
  if (threadIdx.x == 0) atomicAdd(&ws[2], red[0]);
}

// ---------------- per-assigned-cell xy / wh / cls losses ----------------
__global__ void assigned_kernel(const float* __restrict__ xpost, float* ws) {
  const int idx = blockIdx.x * blockDim.x + threadIdx.x;
  if (idx >= NPAIR) return;
  const int b = idx / NBOX, n = idx % NBOX;
  const int* pi = (const int*)(ws + WS_PI);
  const float* pf = ws + WS_PF;
  if (!pi[PI_ASSIGN * NPAIR + idx]) return;
  const int a  = pi[PI_A  * NPAIR + idx];
  const int gi = pi[PI_GI * NPAIR + idx];
  const int gj = pi[PI_GJ * NPAIR + idx];
  // last-write-wins: skip if a later GT scatters to the same (b,a,gj,gi)
  for (int n2 = n + 1; n2 < NBOX; ++n2) {
    int idx2 = b * NBOX + n2;
    if (pi[PI_ASSIGN * NPAIR + idx2] && pi[PI_A * NPAIR + idx2] == a &&
        pi[PI_GI * NPAIR + idx2] == gi && pi[PI_GJ * NPAIR + idx2] == gj) return;
  }
  const int s = gj * NG + gi;
  const size_t base = ((size_t)b * NOUT + a * NCH) * NS + s;
  const float tx = pf[PF_TX * NPAIR + idx], ty = pf[PF_TY * NPAIR + idx];
  const float tw = pf[PF_TW * NPAIR + idx], th = pf[PF_TH * NPAIR + idx];
  const float sv = pf[PF_SV * NPAIR + idx];
  const int  cls = pi[PI_CLS * NPAIR + idx];

  const float p0 = xpost[base];
  const float p1 = xpost[base + (size_t)NS];
  float lxy = -(tx * slogf_(p0) + (1.0f - tx) * slogf_(1.0f - p0))
            - (ty * slogf_(p1) + (1.0f - ty) * slogf_(1.0f - p1));
  lxy *= sv;

  const float w0 = xpost[base + 2 * (size_t)NS];
  const float w1 = xpost[base + 3 * (size_t)NS];
  const float lwh = 0.5f * sv * ((w0 - tw) * (w0 - tw) + (w1 - th) * (w1 - th));

  float lcls = 0.0f;
  for (int c = 0; c < 80; ++c) {
    float p = xpost[base + (size_t)(5 + c) * NS];
    float t = (c == cls) ? 1.0f : 0.0f;
    lcls += -(t * slogf_(p) + (1.0f - t) * slogf_(1.0f - p));
  }
  atomicAdd(&ws[0], lxy);
  atomicAdd(&ws[1], lwh);
  atomicAdd(&ws[3], lcls);
}

// ---------------- finalize ----------------
__global__ void finalize_kernel(const float* ws, float* out) {
  if (threadIdx.x == 0 && blockIdx.x == 0) {
    float lxy = ws[0], lwh = ws[1], lobj = ws[2], lcls = ws[3];
    out[0] = lxy + lwh + lobj + lcls;
    out[1] = lxy;
    out[2] = lwh;
    out[3] = lobj;
    out[4] = lcls;
  }
}

// ---------------- launcher ----------------
extern "C" void kernel_launch(void* const* d_in, const int* in_sizes, int n_in,
                              void* d_out, int out_size, void* d_ws, size_t ws_size,
                              hipStream_t stream) {
  const float* xin    = (const float*)d_in[0];
  const float* labels = (const float*)d_in[1];
  const float* Wm     = (const float*)d_in[2];
  const float* bias   = (const float*)d_in[3];
  float* ws    = (float*)d_ws;
  float* out   = (float*)d_out;
  __bf16* wbf  = (__bf16*)(ws + WS_WBF);
  float* xpost = ws + WS_XPOST;

  hipLaunchKernelGGL(init_kernel, dim3(1), dim3(32), 0, stream, ws);

  hipLaunchKernelGGL(packW_kernel, dim3(16), dim3(256), 0, stream, Wm, wbf);

  // one block per (batch, sTile): 16 * 361 = 5776 blocks, 256 threads
  hipLaunchKernelGGL(gemm_wmma_kernel, dim3(NBATCH * 361), dim3(256), 0, stream,
                     xin, wbf, bias, xpost);

  hipLaunchKernelGGL(prep_kernel, dim3((NPAIR + 255) / 256), dim3(256), 0, stream,
                     labels, ws);

  // per batch: NANCH*NS = 17328 cells -> 68 blocks of 256
  hipLaunchKernelGGL(obj_kernel, dim3((NANCH * NS + 255) / 256, NBATCH), dim3(256), 0, stream,
                     xpost, ws);

  hipLaunchKernelGGL(assigned_kernel, dim3((NPAIR + 255) / 256), dim3(256), 0, stream,
                     xpost, ws);

  hipLaunchKernelGGL(finalize_kernel, dim3(1), dim3(32), 0, stream, ws, out);
}
